// SelfAttention1D_30305289241200
// MI455X (gfx1250) — matrix operand order repaired and checked by reading, not compiled, not run on hardware
//
#include <hip/hip_runtime.h>

#define BB 8
#define CC 256
#define NN 2048
#define NT (NN / 16)      // 128 tiles along N
#define SCALE 0.0625f     // 1/sqrt(256)

typedef __attribute__((ext_vector_type(16))) _Float16 v16h;
typedef __attribute__((ext_vector_type(8)))  _Float16 v8h;
typedef __attribute__((ext_vector_type(4)))  _Float16 v4h;
typedef __attribute__((ext_vector_type(8)))  float    v8f;

__device__ __forceinline__ v16h pack16(v8h lo, v8h hi) {
  v16h a;
#pragma unroll
  for (int i = 0; i < 8; ++i) { a[i] = lo[i]; a[i + 8] = hi[i]; }
  return a;
}

__device__ __forceinline__ v8h ld8h(const _Float16* p) {
  return *reinterpret_cast<const v8h*>(p);
}

// ---------------------------------------------------------------------------
// Kernel 0: one-shot conversion of Wq|Wk|Wv (fp32) -> Wh (f16), 3*C*C elems.
// ---------------------------------------------------------------------------
__global__ __launch_bounds__(256) void convert_w_kernel(
    const float* __restrict__ Wq, const float* __restrict__ Wk,
    const float* __restrict__ Wv, _Float16* __restrict__ Wh)
{
  const int idx = blockIdx.x * 256 + threadIdx.x;   // 4 elements per thread
  const int e   = idx * 4;
  const int mat = e / (CC * CC);
  const int off = e % (CC * CC);
  const float* src = (mat == 0) ? Wq : (mat == 1) ? Wk : Wv;
  float4 f = reinterpret_cast<const float4*>(src + off)[0];
  v4h h;
  h[0] = (_Float16)f.x; h[1] = (_Float16)f.y;
  h[2] = (_Float16)f.z; h[3] = (_Float16)f.w;
  *reinterpret_cast<v4h*>(Wh + (size_t)mat * CC * CC + off) = h;
}

// ---------------------------------------------------------------------------
// Kernel 1: Q/K/V pointwise projections.
// Q,K stored transposed [B,N,C] (f16); V stored natural [B,C,N] (f16).
// ---------------------------------------------------------------------------
__global__ __launch_bounds__(256) void qkv_proj_kernel(
    const float* __restrict__ x, const _Float16* __restrict__ Wh,
    const float* __restrict__ bq, const float* __restrict__ bk,
    const float* __restrict__ bv,
    _Float16* __restrict__ Qt, _Float16* __restrict__ Kt,
    _Float16* __restrict__ Vh)
{
  __shared__ __align__(16) _Float16 xs[16 * CC];   // xs[n][c]  (transposed x tile)

  const int b  = blockIdx.x / NT;
  const int n0 = (blockIdx.x % NT) * 16;
  const int t  = threadIdx.x;

  // cooperative load + transpose of the x tile into LDS (f32 -> f16)
  {
    const int c = t;                                // 256 threads = 256 channels
    const float* xp = x + ((size_t)b * CC + c) * NN + n0;
#pragma unroll
    for (int i = 0; i < 4; ++i) {
      float4 f = reinterpret_cast<const float4*>(xp)[i];
      xs[(4 * i + 0) * CC + c] = (_Float16)f.x;
      xs[(4 * i + 1) * CC + c] = (_Float16)f.y;
      xs[(4 * i + 2) * CC + c] = (_Float16)f.z;
      xs[(4 * i + 3) * CC + c] = (_Float16)f.w;
    }
  }
  __syncthreads();

  const int w  = t >> 5;     // wave 0..7
  const int l  = t & 31;
  const int hf = l >> 4;     // half-wave
  const int lc = l & 15;

#pragma unroll
  for (int j = 0; j < 3; ++j) {
    const _Float16* W  = Wh + (size_t)j * CC * CC;  // f16 weights (L2 resident)
    const float* bias  = (j == 0) ? bq : (j == 1) ? bk : bv;
#pragma unroll
    for (int ti = 0; ti < 2; ++ti) {
      const int m0 = (w * 2 + ti) * 16;            // output-channel tile
      const int m  = m0 + lc;                      // A-matrix row for this lane
      const _Float16* wrow = W + (size_t)m * CC;
      v8f acc = {};
#pragma unroll
      for (int kk = 0; kk < 8; ++kk) {
        const int k0 = kk * 32;
        // A: 16x32 tile of W  (rows = out channels, k = in channels)
        v16h a = pack16(ld8h(wrow + k0 + hf * 16),
                        ld8h(wrow + k0 + hf * 16 + 8));
        // B: 32x16 tile of x  (k = in channels, cols = n); contiguous 16-k run
        const _Float16* bp = &xs[lc * CC + k0 + hf * 16];
        v16h bm = pack16(ld8h(bp), ld8h(bp + 8));
        acc = __builtin_amdgcn_wmma_f32_16x16x32_f16(false, a, false, bm,
                                                     (short)0, acc, false, false);
      }
#pragma unroll
      for (int r = 0; r < 8; ++r) {
        const int ch = m0 + r + 8 * hf;            // C/D row -> output channel
        const int n  = n0 + lc;                    // C/D col -> position
        const float v = acc[r] + bias[ch];
        if (j == 0)      Qt[((size_t)b * NN + n) * CC + ch] = (_Float16)v;
        else if (j == 1) Kt[((size_t)b * NN + n) * CC + ch] = (_Float16)v;
        else             Vh[((size_t)b * CC + ch) * NN + n] = (_Float16)v;
      }
    }
  }
}

// ---------------------------------------------------------------------------
// Kernel 2: per-row softmax statistics (flash pass 1): M[b,n], L[b,n].
// ---------------------------------------------------------------------------
__global__ __launch_bounds__(256) void attn_stats_kernel(
    const _Float16* __restrict__ Qt, const _Float16* __restrict__ Kt,
    float* __restrict__ Mg, float* __restrict__ Lg)
{
  __shared__ __align__(16) _Float16 qs[16 * CC];   // qs[q][c]
  __shared__ float mred[8][16];
  __shared__ float lred[8][16];

  const int b  = blockIdx.x / NT;
  const int n0 = (blockIdx.x % NT) * 16;
  const int t  = threadIdx.x;

  { // load Q tile (already transposed in memory)
    const int q  = t >> 4;
    const int c0 = (t & 15) * 16;
    const _Float16* src = Qt + ((size_t)b * NN + n0 + q) * CC + c0;
    reinterpret_cast<v8h*>(&qs[q * CC + c0])[0] = reinterpret_cast<const v8h*>(src)[0];
    reinterpret_cast<v8h*>(&qs[q * CC + c0])[1] = reinterpret_cast<const v8h*>(src)[1];
  }
  __syncthreads();

  const int w = t >> 5, l = t & 31, hf = l >> 4, lc = l & 15;

  float mrow[8], lrow[8];
#pragma unroll
  for (int r = 0; r < 8; ++r) { mrow[r] = -1e30f; lrow[r] = 0.f; }

  for (int mt = w; mt < NT; mt += 8) {
    const int m0 = mt * 16;
    const _Float16* kp = Kt + ((size_t)b * NN + m0 + lc) * CC;
    if (mt + 8 < NT)   // prefetch next K tile row into caches
      __builtin_prefetch(kp + 8 * 16 * CC, 0, 1);
    v8f s = {};
#pragma unroll
    for (int kk = 0; kk < 8; ++kk) {
      const int k0 = kk * 32;
      v16h a = pack16(ld8h(&qs[lc * CC + k0 + hf * 16]),
                      ld8h(&qs[lc * CC + k0 + hf * 16 + 8]));
      v16h bm = pack16(ld8h(kp + k0 + hf * 16), ld8h(kp + k0 + hf * 16 + 8));
      s = __builtin_amdgcn_wmma_f32_16x16x32_f16(false, a, false, bm,
                                                 (short)0, s, false, false);
    }
#pragma unroll
    for (int r = 0; r < 8; ++r) {
      float v = s[r] * SCALE;
      float rmax = v;                               // row spread over 16 lanes of half
      rmax = fmaxf(rmax, __shfl_xor(rmax, 1));
      rmax = fmaxf(rmax, __shfl_xor(rmax, 2));
      rmax = fmaxf(rmax, __shfl_xor(rmax, 4));
      rmax = fmaxf(rmax, __shfl_xor(rmax, 8));
      const float newm = fmaxf(mrow[r], rmax);
      float e = __expf(v - newm);
      e += __shfl_xor(e, 1);
      e += __shfl_xor(e, 2);
      e += __shfl_xor(e, 4);
      e += __shfl_xor(e, 8);
      lrow[r] = lrow[r] * __expf(mrow[r] - newm) + e;
      mrow[r] = newm;
    }
  }

  if (lc == 0) {
#pragma unroll
    for (int r = 0; r < 8; ++r) {
      mred[w][r + 8 * hf] = mrow[r];
      lred[w][r + 8 * hf] = lrow[r];
    }
  }
  __syncthreads();
  if (t < 16) {
    float M = -1e30f;
#pragma unroll
    for (int ww = 0; ww < 8; ++ww) M = fmaxf(M, mred[ww][t]);
    float L = 0.f;
#pragma unroll
    for (int ww = 0; ww < 8; ++ww) L += lred[ww][t] * __expf(mred[ww][t] - M);
    Mg[(size_t)b * NN + n0 + t] = M;
    Lg[(size_t)b * NN + n0 + t] = L;
  }
}

// ---------------------------------------------------------------------------
// Kernel 3: O = softmax(S) * V^T  (flash pass 2), fused residual +x.
// ---------------------------------------------------------------------------
__global__ __launch_bounds__(256) void attn_out_kernel(
    const float* __restrict__ x,
    const _Float16* __restrict__ Qt, const _Float16* __restrict__ Kt,
    const _Float16* __restrict__ Vh,
    const float* __restrict__ Mg, const float* __restrict__ Lg,
    float* __restrict__ out)
{
  __shared__ __align__(16) _Float16 qs[16 * CC];   // qs[q][c]
  __shared__ __align__(16) _Float16 ps[16 * 128];  // P tile: [q][m in 128-chunk]
  __shared__ float Ml[16], Linv[16];

  const int b  = blockIdx.x / NT;
  const int n0 = (blockIdx.x % NT) * 16;
  const int t  = threadIdx.x;

  { // load Q tile
    const int q  = t >> 4;
    const int c0 = (t & 15) * 16;
    const _Float16* src = Qt + ((size_t)b * NN + n0 + q) * CC + c0;
    reinterpret_cast<v8h*>(&qs[q * CC + c0])[0] = reinterpret_cast<const v8h*>(src)[0];
    reinterpret_cast<v8h*>(&qs[q * CC + c0])[1] = reinterpret_cast<const v8h*>(src)[1];
  }
  if (t < 16) {
    Ml[t]   = Mg[(size_t)b * NN + n0 + t];
    Linv[t] = 1.0f / Lg[(size_t)b * NN + n0 + t];
  }
  __syncthreads();

  const int w = t >> 5, l = t & 31, hf = l >> 4, lc = l & 15;
  const int c0 = w * 32;                           // this wave's output channels

  v8f o0 = {}, o1 = {};

  for (int outer = 0; outer < NT / 8; ++outer) {   // 16 iterations x 128 keys
    const int mb = outer * 128;

    // --- this wave computes P for its 16-key tile ---
    const int m0 = mb + w * 16;
    const _Float16* kp = Kt + ((size_t)b * NN + m0 + lc) * CC;
    if (outer + 1 < NT / 8)  // prefetch next iteration's K tile row
      __builtin_prefetch(kp + (size_t)128 * CC, 0, 1);
    v8f s = {};
#pragma unroll
    for (int kk = 0; kk < 8; ++kk) {
      const int k0 = kk * 32;
      v16h a = pack16(ld8h(&qs[lc * CC + k0 + hf * 16]),
                      ld8h(&qs[lc * CC + k0 + hf * 16 + 8]));
      v16h bm = pack16(ld8h(kp + k0 + hf * 16), ld8h(kp + k0 + hf * 16 + 8));
      s = __builtin_amdgcn_wmma_f32_16x16x32_f16(false, a, false, bm,
                                                 (short)0, s, false, false);
    }
#pragma unroll
    for (int r = 0; r < 8; ++r) {
      const int q = r + 8 * hf;
      const float p = __expf(s[r] * SCALE - Ml[q]) * Linv[q];
      ps[q * 128 + w * 16 + lc] = (_Float16)p;
    }
    __syncthreads();

    // --- all waves: O += P(16x128-chunk) x V^T for their channel slice ---
#pragma unroll
    for (int kt = 0; kt < 4; ++kt) {
      const int k0 = kt * 32;
      v16h a = pack16(ld8h(&ps[lc * 128 + k0 + hf * 16]),
                      ld8h(&ps[lc * 128 + k0 + hf * 16 + 8]));
      const _Float16* vp0 = Vh + ((size_t)b * CC + c0 + lc) * NN + mb + k0 + hf * 16;
      v16h b0 = pack16(ld8h(vp0), ld8h(vp0 + 8));
      o0 = __builtin_amdgcn_wmma_f32_16x16x32_f16(false, a, false, b0,
                                                  (short)0, o0, false, false);
      const _Float16* vp1 = Vh + ((size_t)b * CC + c0 + 16 + lc) * NN + mb + k0 + hf * 16;
      v16h b1 = pack16(ld8h(vp1), ld8h(vp1 + 8));
      o1 = __builtin_amdgcn_wmma_f32_16x16x32_f16(false, a, false, b1,
                                                  (short)0, o1, false, false);
    }
    __syncthreads();
  }

  // out = O + x  (residual), scatter per C/D layout
#pragma unroll
  for (int r = 0; r < 8; ++r) {
    const int q = r + 8 * hf;
    const size_t i0 = ((size_t)b * CC + (c0 + lc)) * NN + (n0 + q);
    out[i0] = o0[r] + x[i0];
    const size_t i1 = ((size_t)b * CC + (c0 + 16 + lc)) * NN + (n0 + q);
    out[i1] = o1[r] + x[i1];
  }
}

// ---------------------------------------------------------------------------
extern "C" void kernel_launch(void* const* d_in, const int* in_sizes, int n_in,
                              void* d_out, int out_size, void* d_ws, size_t ws_size,
                              hipStream_t stream) {
  (void)in_sizes; (void)n_in; (void)out_size; (void)ws_size;
  const float* x  = (const float*)d_in[0];
  const float* Wq = (const float*)d_in[1];
  const float* bq = (const float*)d_in[2];
  const float* Wk = (const float*)d_in[3];
  const float* bk = (const float*)d_in[4];
  const float* Wv = (const float*)d_in[5];
  const float* bv = (const float*)d_in[6];
  float* out = (float*)d_out;

  // workspace: Qt,Kt [B,N,C] f16 ; V [B,C,N] f16 ; M,L [B,N] f32 ; Wh [3,C,C] f16
  _Float16* Qt = (_Float16*)d_ws;
  _Float16* Kt = Qt + (size_t)BB * NN * CC;
  _Float16* Vh = Kt + (size_t)BB * NN * CC;
  float*    Mg = (float*)(Vh + (size_t)BB * NN * CC);
  float*    Lg = Mg + (size_t)BB * NN;
  _Float16* Wh = (_Float16*)(Lg + (size_t)BB * NN);

  dim3 block(256);      // 8 waves (wave32)

  convert_w_kernel<<<dim3(3 * CC * CC / (256 * 4)), block, 0, stream>>>(Wq, Wk, Wv, Wh);
  qkv_proj_kernel<<<dim3(BB * NT), block, 0, stream>>>(x, Wh, bq, bk, bv, Qt, Kt, Vh);
  attn_stats_kernel<<<dim3(BB * NT), block, 0, stream>>>(Qt, Kt, Mg, Lg);
  attn_out_kernel<<<dim3(BB * NT), block, 0, stream>>>(x, Qt, Kt, Vh, Mg, Lg, out);
}
